// VanillaRNN_80908593923381
// MI455X (gfx1250) — matrix operand order repaired and checked by reading, not compiled, not run on hardware
//
#include <hip/hip_runtime.h>

// ---------------------------------------------------------------------------
// Vanilla RNN on MI455X (gfx1250, wave32):
//   h_{t} = tanh(Whh @ h_{t-1} + whx * x_t + bh)   (128 sequential steps)
//   out   = Wph @ h_S + bp
// Core: per-step 2048x512x2048 GEMM via v_wmma_f32_16x16x32_f16.
// Global->LDS staging uses the CDNA5 async DMA path
// (global_load_async_to_lds_b128, ASYNCcnt) with double-buffered LDS and
// static buffer addresses, so no VGPR staging and no loadcnt/ds_store chain.
// h kept transposed as hT[batch][hidden] (fp16) so A/B fragments and C stores
// are contiguous per-lane per the CDNA5 WMMA VGPR layouts.
// ---------------------------------------------------------------------------

typedef __attribute__((ext_vector_type(16))) _Float16 v16h;
typedef __attribute__((ext_vector_type(8)))  _Float16 v8h;
typedef __attribute__((ext_vector_type(8)))  float    v8f;
typedef __attribute__((ext_vector_type(4)))  float    v4f;

#define HID 2048
#define BAT 512
#define SEQ 128
#define NCLS 10

#define BM 64
#define BN 64
#define BK 64   // k-chunk per LDS stage (2 WMMA k-substeps of 32)

// --------------------------- fp32 -> fp16 convert ---------------------------
__global__ void rnn_cvt_f16(const float* __restrict__ src,
                            _Float16* __restrict__ dst, int n) {
    int i = blockIdx.x * blockDim.x + threadIdx.x;
    if (i < n) dst[i] = (_Float16)src[i];
}

// --------------------------- step 0 (h_prev == 0) ---------------------------
__global__ void rnn_step0(const float* __restrict__ whx,
                          const float* __restrict__ bh,
                          const float* __restrict__ x,
                          _Float16* __restrict__ h) {
    int i = blockIdx.x * blockDim.x + threadIdx.x;   // over BAT*HID
    int b = i >> 11;
    int m = i & (HID - 1);
    float v = tanhf(whx[m] * x[b * SEQ + 0] + bh[m]);
    h[(size_t)b * HID + m] = (_Float16)v;
}

// ------------------------------ RNN GEMM step -------------------------------
// hnext[n][m] = tanh( sum_k Whh[m][k]*hprev[n][k] + whx[m]*x[n][t] + bh[m] )

// Async DMA: 64B per thread per tile, INST_OFFSET advances LDS and global
// addresses together (CDNA5 ISA).
__device__ __forceinline__ void async_stage(unsigned ldsA, unsigned ldsB,
                                            const _Float16* ga,
                                            const _Float16* gb) {
    asm volatile(
        "global_load_async_to_lds_b128 %0, %2, off\n\t"
        "global_load_async_to_lds_b128 %0, %2, off offset:16\n\t"
        "global_load_async_to_lds_b128 %0, %2, off offset:32\n\t"
        "global_load_async_to_lds_b128 %0, %2, off offset:48\n\t"
        "global_load_async_to_lds_b128 %1, %3, off\n\t"
        "global_load_async_to_lds_b128 %1, %3, off offset:16\n\t"
        "global_load_async_to_lds_b128 %1, %3, off offset:32\n\t"
        "global_load_async_to_lds_b128 %1, %3, off offset:48"
        :
        : "v"(ldsA), "v"(ldsB), "v"(ga), "v"(gb)
        : "memory");
}

__device__ __forceinline__ void async_wait_sync() {
    asm volatile("s_wait_asynccnt 0x0" ::: "memory");
    __syncthreads();
}

__device__ __forceinline__ void compute_tile(const _Float16* __restrict__ As,
                                             const _Float16* __restrict__ Bs,
                                             int wm, int wn, int l16, int lh,
                                             v8f acc[2][2]) {
#pragma unroll
    for (int ks = 0; ks < 2; ++ks) {        // two 16x16x32 k-substeps
        const int kOff = ks * 32;
        union { v16h v; v8h h[2]; } a[2], b[2];
        // A-fragment (16x32): lanes<16 -> K 0..7 & 16..23; lanes>=16 -> +8
#pragma unroll
        for (int mi = 0; mi < 2; ++mi) {
            int r = wm * 32 + mi * 16 + l16;
            a[mi].h[0] = *(const v8h*)&As[r * BK + kOff + lh * 8];
            a[mi].h[1] = *(const v8h*)&As[r * BK + kOff + lh * 8 + 16];
        }
        // B-fragment (32x16): lane = column n; lanes<16 -> K 0..7 & 16..23; lanes>=16 -> +8
#pragma unroll
        for (int ni = 0; ni < 2; ++ni) {
            int c = wn * 32 + ni * 16 + l16;
            b[ni].h[0] = *(const v8h*)&Bs[c * BK + kOff + lh * 8];
            b[ni].h[1] = *(const v8h*)&Bs[c * BK + kOff + lh * 8 + 16];
        }
#pragma unroll
        for (int mi = 0; mi < 2; ++mi)
#pragma unroll
            for (int ni = 0; ni < 2; ++ni)
                acc[mi][ni] = __builtin_amdgcn_wmma_f32_16x16x32_f16(
                    false, a[mi].v, false, b[ni].v,
                    (short)0, acc[mi][ni], false, false);
    }
}

__global__ __launch_bounds__(128)
void rnn_step(const _Float16* __restrict__ Whh,    // [HID][HID] row-major f16
              const _Float16* __restrict__ hprev,  // [BAT][HID] f16
              _Float16* __restrict__ hnext,        // [BAT][HID] f16
              const float* __restrict__ whx,       // [HID]
              const float* __restrict__ bh,        // [HID]
              const float* __restrict__ x,         // [BAT][SEQ]
              int t) {
    __shared__ __align__(16) _Float16 As[2][BM * BK];   // 2 x 8KB
    __shared__ __align__(16) _Float16 Bs[2][BN * BK];   // 2 x 8KB

    const int tid  = threadIdx.x;       // 0..127 (4 waves of 32)
    const int lane = tid & 31;
    const int wave = tid >> 5;
    const int wm   = wave & 1;          // wave tile row (2 x 32 rows)
    const int wn   = wave >> 1;         // wave tile col (2 x 32 cols)
    const int mBlock = blockIdx.x * BM; // 2048/64 = 32
    const int nBlock = blockIdx.y * BN; //  512/64 = 8

    const int l16 = lane & 15;
    const int lh  = lane >> 4;          // half-wave select

    v8f acc[2][2] = {};                 // [mi][ni] 16x16 f32 accumulators

    // staging geometry: 64 rows x 64 k-elems (128B/row), 2 threads per row,
    // each thread DMAs 64B per tile
    const int lrow  = tid >> 1;             // 0..63
    const int lhalf = (tid & 1) * 32;       // f16 element offset 0 or 32
    const int soff  = lrow * BK + lhalf;    // element offset inside a tile
    const _Float16* Ag = Whh   + (size_t)(mBlock + lrow) * HID + lhalf;
    const _Float16* Bg = hprev + (size_t)(nBlock + lrow) * HID + lhalf;

    // LDS byte addresses (flat->LDS truncation per ISA: low 32 bits = offset)
    const unsigned ldsA0 = (unsigned)(size_t)&As[0][soff];
    const unsigned ldsA1 = (unsigned)(size_t)&As[1][soff];
    const unsigned ldsB0 = (unsigned)(size_t)&Bs[0][soff];
    const unsigned ldsB1 = (unsigned)(size_t)&Bs[1][soff];

    // prologue: DMA chunk 0 into buffer 0
    async_stage(ldsA0, ldsB0, Ag, Bg);
    async_wait_sync();

    // main loop, unrolled x2 with static buffer addresses.
    // Invariant at loop top: buf0 holds chunk (kc-BK)/BK.
    int kc = BK;
    for (; kc < HID - BK; kc += 2 * BK) {
        async_stage(ldsA1, ldsB1, Ag + kc, Bg + kc);            // chunk j -> buf1
        compute_tile(&As[0][0], &Bs[0][0], wm, wn, l16, lh, acc); // chunk j-1
        async_wait_sync();
        async_stage(ldsA0, ldsB0, Ag + kc + BK, Bg + kc + BK);  // chunk j+1 -> buf0
        compute_tile(&As[1][0], &Bs[1][0], wm, wn, l16, lh, acc); // chunk j
        async_wait_sync();
    }
    // tail: kc == HID-BK; buf0 holds chunk 30, DMA chunk 31 into buf1
    async_stage(ldsA1, ldsB1, Ag + kc, Bg + kc);
    compute_tile(&As[0][0], &Bs[0][0], wm, wn, l16, lh, acc);
    async_wait_sync();
    compute_tile(&As[1][0], &Bs[1][0], wm, wn, l16, lh, acc);

    // Epilogue: C element r of lane L -> m = base + (L/16)*8 + r, n = base + L%16
#pragma unroll
    for (int mi = 0; mi < 2; ++mi) {
        int m0 = mBlock + wm * 32 + mi * 16 + lh * 8;  // 8 consecutive rows
        v4f wx0 = *(const v4f*)(whx + m0);
        v4f wx1 = *(const v4f*)(whx + m0 + 4);
        v4f bb0 = *(const v4f*)(bh + m0);
        v4f bb1 = *(const v4f*)(bh + m0 + 4);
#pragma unroll
        for (int ni = 0; ni < 2; ++ni) {
            int n = nBlock + wn * 32 + ni * 16 + l16;   // batch index
            float xv = x[n * SEQ + t];
            v8h out;
#pragma unroll
            for (int r = 0; r < 8; ++r) {
                float w  = (r < 4) ? wx0[r] : wx1[r - 4];
                float bb = (r < 4) ? bb0[r] : bb1[r - 4];
                out[r] = (_Float16)tanhf(acc[mi][ni][r] + w * xv + bb);
            }
            *(v8h*)(hnext + (size_t)n * HID + m0) = out;  // 16B contiguous
        }
    }
}

// ---------------------------- final projection ------------------------------
__global__ void rnn_proj(const float* __restrict__ wph,
                         const float* __restrict__ bp,
                         const _Float16* __restrict__ h,
                         float* __restrict__ out) {
    int i = blockIdx.x * blockDim.x + threadIdx.x;
    if (i >= BAT * NCLS) return;
    int b = i / NCLS, c = i % NCLS;
    const _Float16* hb = h + (size_t)b * HID;
    const float* w = wph + (size_t)c * HID;
    float s = 0.f;
    for (int k = 0; k < HID; k += 8) {
        v8h hv = *(const v8h*)(hb + k);
#pragma unroll
        for (int j = 0; j < 8; ++j) s += w[k + j] * (float)hv[j];
    }
    out[b * NCLS + c] = s + bp[c];
}

// --------------------------------- launch -----------------------------------
extern "C" void kernel_launch(void* const* d_in, const int* in_sizes, int n_in,
                              void* d_out, int out_size, void* d_ws, size_t ws_size,
                              hipStream_t stream) {
    const float* x   = (const float*)d_in[0];   // [512][128]
    const float* whx = (const float*)d_in[1];   // [2048]
    const float* whh = (const float*)d_in[2];   // [2048][2048]
    const float* bh  = (const float*)d_in[3];   // [2048]
    const float* wph = (const float*)d_in[4];   // [10][2048]
    const float* bp  = (const float*)d_in[5];   // [10]
    float* out = (float*)d_out;                 // [512][10]

    char* ws = (char*)d_ws;
    _Float16* Wf16 = (_Float16*)ws;                              // 8 MB
    _Float16* hA   = (_Float16*)(ws + (size_t)HID * HID * 2);    // 2 MB
    _Float16* hB   = hA + (size_t)BAT * HID;                     // 2 MB

    rnn_cvt_f16<<<(HID * HID) / 256, 256, 0, stream>>>(whh, Wf16, HID * HID);
    rnn_step0<<<(BAT * HID) / 256, 256, 0, stream>>>(whx, bh, x, hA);

    _Float16* cur = hA;
    _Float16* nxt = hB;
    dim3 grid(HID / BM, BAT / BN);   // 32 x 8 = 256 workgroups
    for (int t = 1; t < SEQ; ++t) {
        rnn_step<<<grid, 128, 0, stream>>>(Wf16, cur, nxt, whx, bh, x, t);
        _Float16* tmp = cur; cur = nxt; nxt = tmp;
    }

    rnn_proj<<<(BAT * NCLS + 255) / 256, 256, 0, stream>>>(wph, bp, cur, out);
}
